// My_lstm_30099130810548
// MI455X (gfx1250) — matrix operand order repaired and checked by reading, not compiled, not run on hardware
//
#include <hip/hip_runtime.h>

// ---------------------------------------------------------------------------
// 2-layer LSTM (B=256, T=1024, E=128, U=256) for gfx1250 (MI455X).
// Batch-split persistent workgroups: 16 blocks x 512 threads (16 wave32
// waves). Each block owns 16 batch rows and runs the full 1024-step
// recurrence locally. Matrix work uses v_wmma_f32_16x16x32_bf16 with f32
// accumulation; weights are converted once to transposed bf16 in d_ws and
// stay resident in the 192MB L2. Weight loads go through explicit
// address_space(1) pointers (pure global_load_b128, LOADcnt only) and the
// base pointers are laundered per timestep so the compiler cannot
// LICM-hoist + spill the fragments.
// ---------------------------------------------------------------------------

typedef __attribute__((ext_vector_type(16))) __bf16 bf16x16;
typedef __attribute__((ext_vector_type(8)))  __bf16 bf16x8;
typedef __attribute__((ext_vector_type(8)))  float  f32x8;

// Global (address_space(1)) pointer types: force global_load codegen.
typedef const __attribute__((address_space(1))) __bf16*  gwp;
typedef const __attribute__((address_space(1))) bf16x8*  gv8p;

namespace {

constexpr int kB = 256;    // batch
constexpr int kT = 1024;   // time steps
constexpr int kE = 128;    // embedding dim
constexpr int kU = 256;    // hidden units
constexpr int kG = 4 * kU; // 1024 gate columns
constexpr int kRows = 16;  // batch rows per block
constexpr int kThreads = 512;           // 16 waves
constexpr int kTilesPerWave = 4;        // 64 gate columns per wave

__device__ __forceinline__ float sigm(float x) {
  return 1.0f / (1.0f + __expf(-x));
}

// tanh via the hardware exp unit (v_exp_f32) instead of libm tanhf.
__device__ __forceinline__ float fast_tanh(float x) {
  float e = __expf(-2.0f * x);
  return (1.0f - e) / (1.0f + e);
}

// Defeat LICM: make the (SGPR, global-AS) pointer value opaque each timestep
// so weight-fragment loads stay inside the loop (L2 hits) instead of being
// hoisted and spilled.
__device__ __forceinline__ gwp launder(gwp p) {
  asm volatile("" : "+s"(p));
  return p;
}

// C/D init: every VGPR of a lane sits in the same column N = nBase + (lane&15),
// so the bias for that column splats across the 8 accumulator registers.
__device__ __forceinline__ f32x8 bias_init(const float* __restrict__ b,
                                           int nBase, int lane) {
  float v = b[nBase + (lane & 15)];
  f32x8 a;
#pragma unroll
  for (int i = 0; i < 8; ++i) a[i] = v;
  return a;
}

// A fragment (16x32 bf16, row-major staging in LDS, stride ld):
// lane&15 = row M; lanes 0-15 hold K = kb+0..7 and kb+16..23,
// lanes 16-31 hold K = kb+8..15 and kb+24..31 (ISA 16-bit A 16x32 layout).
__device__ __forceinline__ bf16x16 load_a_frag(const __bf16* __restrict__ Am,
                                               int ld, int kBase, int lane) {
  int m  = lane & 15;
  int k0 = kBase + ((lane >> 4) << 3);
  bf16x8 lo = *(const bf16x8*)(Am + m * ld + k0);
  bf16x8 hi = *(const bf16x8*)(Am + m * ld + k0 + 16);
  bf16x16 r;
#pragma unroll
  for (int i = 0; i < 8; ++i) { r[i] = lo[i]; r[i + 8] = hi[i]; }
  return r;
}

// B fragment (32x16 bf16). Weights stored transposed [N][K] so each lane's
// K-values are contiguous: lanes 0-15 hold K = kb+0..7 and kb+16..23 at N = nBase+lane,
// lanes 16-31 hold K = kb+8..15 and kb+24..31 (matches A fragment K order).
__device__ __forceinline__ bf16x16 load_b_frag(gwp Bw, int ld, int nBase,
                                               int kBase, int lane) {
  int n  = nBase + (lane & 15);
  int k0 = kBase + ((lane >> 4) << 3);
  bf16x8 lo = *(gv8p)(Bw + n * ld + k0);
  bf16x8 hi = *(gv8p)(Bw + n * ld + k0 + 16);
  bf16x16 r;
#pragma unroll
  for (int i = 0; i < 8; ++i) { r[i] = lo[i]; r[i + 8] = hi[i]; }
  return r;
}

template <int K>
__device__ __forceinline__ f32x8 gemm_tile(f32x8 acc,
                                           const __bf16* __restrict__ Am, int lda,
                                           gwp Bw, int ldb,
                                           int nBase, int lane) {
#pragma unroll
  for (int kb = 0; kb < K; kb += 32) {
    bf16x16 a = load_a_frag(Am, lda, kb, lane);
    bf16x16 b = load_b_frag(Bw, ldb, nBase, kb, lane);
    acc = __builtin_amdgcn_wmma_f32_16x16x32_bf16(
        /*neg_a=*/false, a, /*neg_b=*/false, b,
        /*c_mod=*/(short)0, acc, /*reuse_a=*/false, /*reuse_b=*/false);
  }
  return acc;
}

// D layout: VGPR r -> row M = r + 8*(lane>=16), col N = nBase + (lane&15).
__device__ __forceinline__ void store_tile(float* __restrict__ Z, f32x8 acc,
                                           int nBase, int lane) {
  int n    = nBase + (lane & 15);
  int mOff = (lane >> 4) << 3;
#pragma unroll
  for (int r = 0; r < 8; ++r) Z[(mOff + r) * kG + n] = acc[r];
}

}  // namespace

// ---------------------------------------------------------------------------
// One-shot weight conversion: fp32 [K][4U] row-major -> bf16 [4U][K] in d_ws.
// ---------------------------------------------------------------------------
__global__ void cvt_weights(const float* __restrict__ W0,
                            const float* __restrict__ R0,
                            const float* __restrict__ W1,
                            const float* __restrict__ R1,
                            __bf16* __restrict__ ws) {
  const int TOT0 = kG * kE;  // Wt0
  const int TOT1 = kG * kU;  // Rt0 / Wt1 / Rt1
  const int total = TOT0 + 3 * TOT1;
  for (int idx = blockIdx.x * blockDim.x + threadIdx.x; idx < total;
       idx += gridDim.x * blockDim.x) {
    if (idx < TOT0) {
      int n = idx / kE, k = idx % kE;
      ws[idx] = (__bf16)W0[k * kG + n];
    } else {
      int r = idx - TOT0;
      int sel = r / TOT1, off = r % TOT1;
      int n = off / kU, k = off % kU;
      const float* src = (sel == 0) ? R0 : (sel == 1) ? W1 : R1;
      ws[idx] = (__bf16)src[k * kG + n];
    }
  }
}

// ---------------------------------------------------------------------------
// Persistent recurrence kernel.
// ---------------------------------------------------------------------------
__global__ __launch_bounds__(kThreads, 1)
void lstm2_wmma(const int* __restrict__ tokens, const float* __restrict__ emb,
                const float* __restrict__ b0, const float* __restrict__ b1,
                const float* __restrict__ Wout, const float* __restrict__ bout,
                const __bf16* __restrict__ Wt0g, const __bf16* __restrict__ Rt0g,
                const __bf16* __restrict__ Wt1g, const __bf16* __restrict__ Rt1g,
                float* __restrict__ out) {
  __shared__ __bf16 sX[kRows * kE];    //  4 KB  embedded x_t (bf16)
  __shared__ __bf16 sH0[kRows * kU];   //  8 KB  layer0 hidden (bf16)
  __shared__ __bf16 sH1[kRows * kU];   //  8 KB  layer1 hidden (bf16)
  __shared__ float  sZ[kRows * kG];    // 64 KB  gate pre-activations (f32)

  const int tid     = threadIdx.x;
  const int lane    = tid & 31;
  const int wave    = tid >> 5;          // 0..15
  const int rowBase = blockIdx.x * kRows;

  // Global-AS views of the bf16 weight arrays (pure global_load codegen).
  gwp Wt0b = (gwp)(unsigned long long)Wt0g;
  gwp Rt0b = (gwp)(unsigned long long)Rt0g;
  gwp Wt1b = (gwp)(unsigned long long)Wt1g;
  gwp Rt1b = (gwp)(unsigned long long)Rt1g;

  // h-state zero init; c-state lives in registers (fixed thread->element map).
  for (int i = tid; i < kRows * kU; i += kThreads) {
    sH0[i] = (__bf16)0.0f;
    sH1[i] = (__bf16)0.0f;
  }
  float c0[8], c1[8];
#pragma unroll
  for (int j = 0; j < 8; ++j) { c0[j] = 0.0f; c1[j] = 0.0f; }
  __syncthreads();

#pragma unroll 1
  for (int t = 0; t < kT; ++t) {
    // Opaque copies: keep weight-fragment loads inside the loop.
    gwp Wt0 = launder(Wt0b);
    gwp Rt0 = launder(Rt0b);
    gwp Wt1 = launder(Wt1b);
    gwp Rt1 = launder(Rt1b);

    // --- embedding gather for this step's 16 rows -------------------------
    for (int i = tid; i < kRows * kE; i += kThreads) {
      int r = i >> 7, col = i & (kE - 1);
      int tok = tokens[(rowBase + r) * kT + t];
      sX[i] = (__bf16)emb[tok * kE + col];
    }
    __syncthreads();

    // --- layer 0: z = x@W0 + h0@R0 + b0  (each wave: 4 column tiles, ------
    //     processed one at a time to bound register pressure) --------------
#pragma unroll 1
    for (int j = 0; j < kTilesPerWave; ++j) {
      int nBase = (wave * kTilesPerWave + j) * 16;
      f32x8 acc = bias_init(b0, nBase, lane);
      acc = gemm_tile<kE>(acc, sX,  kE, Wt0, kE, nBase, lane);
      acc = gemm_tile<kU>(acc, sH0, kU, Rt0, kU, nBase, lane);
      store_tile(sZ, acc, nBase, lane);
    }
    __syncthreads();

    // --- layer 0 gates / state update -------------------------------------
#pragma unroll
    for (int j = 0; j < 8; ++j) {
      int idx = tid + j * kThreads;           // 4096 = 16 rows x 256 units
      int r = idx >> 8, u = idx & (kU - 1);
      const float* zr = sZ + r * kG;
      float ii = sigm(zr[u]);
      float ff = sigm(zr[kU + u]);
      float gg = fast_tanh(zr[2 * kU + u]);
      float oo = sigm(zr[3 * kU + u]);
      float c  = ff * c0[j] + ii * gg;
      c0[j] = c;
      sH0[r * kU + u] = (__bf16)(oo * fast_tanh(c));
    }
    __syncthreads();

    // --- layer 1: z = h0@W1 + h1@R1 + b1 ----------------------------------
#pragma unroll 1
    for (int j = 0; j < kTilesPerWave; ++j) {
      int nBase = (wave * kTilesPerWave + j) * 16;
      f32x8 acc = bias_init(b1, nBase, lane);
      acc = gemm_tile<kU>(acc, sH0, kU, Wt1, kU, nBase, lane);
      acc = gemm_tile<kU>(acc, sH1, kU, Rt1, kU, nBase, lane);
      store_tile(sZ, acc, nBase, lane);
    }
    __syncthreads();

    // --- layer 1 gates / state update -------------------------------------
#pragma unroll
    for (int j = 0; j < 8; ++j) {
      int idx = tid + j * kThreads;
      int r = idx >> 8, u = idx & (kU - 1);
      const float* zr = sZ + r * kG;
      float ii = sigm(zr[u]);
      float ff = sigm(zr[kU + u]);
      float gg = fast_tanh(zr[2 * kU + u]);
      float oo = sigm(zr[3 * kU + u]);
      float c  = ff * c1[j] + ii * gg;
      c1[j] = c;
      sH1[r * kU + u] = (__bf16)(oo * fast_tanh(c));
    }
    __syncthreads();
  }

  // --- output head: sigmoid(h1 @ Wout + bout) -----------------------------
  if (tid < kRows) {
    float acc = bout[0];
    const __bf16* h = sH1 + tid * kU;
#pragma unroll 4
    for (int k = 0; k < kU; ++k) acc += (float)h[k] * Wout[k];
    out[rowBase + tid] = sigm(acc);
  }
}

// ---------------------------------------------------------------------------
extern "C" void kernel_launch(void* const* d_in, const int* in_sizes, int n_in,
                              void* d_out, int out_size, void* d_ws,
                              size_t ws_size, hipStream_t stream) {
  const int*   tokens = (const int*)d_in[0];
  const float* emb    = (const float*)d_in[1];
  const float* W0     = (const float*)d_in[2];
  const float* R0     = (const float*)d_in[3];
  const float* b0     = (const float*)d_in[4];
  const float* W1     = (const float*)d_in[5];
  const float* R1     = (const float*)d_in[6];
  const float* b1     = (const float*)d_in[7];
  const float* Wout   = (const float*)d_in[8];
  const float* bout   = (const float*)d_in[9];
  float*       out    = (float*)d_out;

  // Workspace: transposed bf16 weights, 1.75 MB total (L2-resident).
  __bf16* ws  = (__bf16*)d_ws;
  __bf16* Wt0 = ws;
  __bf16* Rt0 = Wt0 + (size_t)kG * kE;
  __bf16* Wt1 = Rt0 + (size_t)kG * kU;
  __bf16* Rt1 = Wt1 + (size_t)kG * kU;

  cvt_weights<<<896, 1024, 0, stream>>>(W0, R0, W1, R1, ws);
  lstm2_wmma<<<kB / kRows, kThreads, 0, stream>>>(
      tokens, emb, b0, b1, Wout, bout, Wt0, Rt0, Wt1, Rt1, out);
}
